// LDS_42305427865777
// MI455X (gfx1250) — matrix operand order repaired and checked by reading, not compiled, not run on hardware
//
#include <hip/hip_runtime.h>
#include <cstdint>

typedef __attribute__((ext_vector_type(16))) _Float16 v16h;
typedef __attribute__((ext_vector_type(8)))  float    v8f;

#define BSZ    256
#define SEQ    512
#define NSTATE 512
#define DOUT   512
#define KXX    5
#define KEXT   544          // 512 state cols + 5 AR taps + 27 zero pad = 17*32
#define KTILES 17
#define TM     128          // M rows per block
#define TN     64           // N cols per block
#define LDA    40           // LDS stride (halves): 80B rows -> 16B aligned, 20-dword stride (conflict-free)
#define LDB    40           // same for B (stored [n][k] in LDS)

// ---------------- Kernel 0: pack extended B matrix TRANSPOSED: Bt[n][k] f16 ----------
// k 0..511 = C[k][n];  k 512..516 = M[n][0][k-512];  k 517..543 = 0
__global__ void pack_b(const float* __restrict__ C, const float* __restrict__ M,
                       _Float16* __restrict__ Bt) {
  int idx = blockIdx.x * blockDim.x + threadIdx.x;
  if (idx >= DOUT * KEXT) return;
  int n = idx / KEXT, k = idx % KEXT;
  float v = 0.0f;
  if (k < NSTATE)            v = C[k * DOUT + n];
  else if (k < NSTATE + KXX) v = M[n * KXX + (k - NSTATE)];
  Bt[idx] = (_Float16)v;    // contiguous writes along k
}

// ---------------- Kernel 1: sequential scan + pack extended A matrix ----------------
// One block per batch element. Threads 0..511: state scan (fp32). Threads 512..516:
// shifted-x AR taps. Threads 517..543: zero pad. Row m = b*SEQ + t, length KEXT, f16.
__global__ __launch_bounds__(KEXT) void scan_pack_a(
    const float* __restrict__ x, const float* __restrict__ A,
    const float* __restrict__ Bv, const float* __restrict__ h0,
    _Float16* __restrict__ Ah) {
  __shared__ float xs[SEQ];
  const int b = blockIdx.x;
  const int tid = threadIdx.x;
  if (tid < SEQ) xs[tid] = x[b * SEQ + tid];
  __syncthreads();

  float h = 0.0f, a = 0.0f, bcoef = 0.0f;
  if (tid < NSTATE) { h = h0[tid]; a = A[tid]; bcoef = Bv[tid]; }

  _Float16* row = Ah + (size_t)b * SEQ * KEXT + tid;
  for (int t = 0; t < SEQ; ++t) {
    float val;
    if (tid < NSTATE) {
      h = h * a + xs[t] * bcoef;      // h_t = A*h_{t-1} + x_t*B
      val = h;
    } else if (tid < NSTATE + KXX) {
      int tt = t - 1 - (tid - NSTATE);
      val = (tt >= 0) ? xs[tt] : 0.0f;
    } else {
      val = 0.0f;
    }
    row[(size_t)t * KEXT] = (_Float16)val;   // coalesced across tid
  }
}

// ---------------- Kernel 2: tiled WMMA GEMM: out[m][o] = A_ext[m][k] * Bt[o][k] ------
// Double-buffered LDS, one barrier per K-step, all tile traffic as b128.
#define LOAD_TILES(kk_, buf_) do {                                               \
    const int k0_ = (kk_) * 32;                                                  \
    { /* A: 128 rows x 32 halves = 8 KB, 2x uint4 per thread */                  \
      const int r_  = tid >> 1;                                                  \
      const int ho_ = (tid & 1) * 16;                                            \
      const uint4* s_ = (const uint4*)(Ah + (m0 + r_) * KEXT + k0_ + ho_);       \
      uint4* d_ = (uint4*)(&ldsA[buf_][r_ * LDA + ho_]);                         \
      d_[0] = s_[0]; d_[1] = s_[1];                                              \
    }                                                                            \
    { /* B: 64 rows x 32 halves = 4 KB, 1x uint4 per thread */                   \
      const int r_  = tid >> 2;                                                  \
      const int qo_ = (tid & 3) * 8;                                             \
      const uint4* s_ = (const uint4*)(Bt + (size_t)(n0 + r_) * KEXT + k0_ + qo_); \
      uint4* d_ = (uint4*)(&ldsB[buf_][r_ * LDB + qo_]);                         \
      d_[0] = s_[0];                                                             \
    }                                                                            \
  } while (0)

__global__ __launch_bounds__(256) void gemm_wmma(
    const _Float16* __restrict__ Ah, const _Float16* __restrict__ Bt,
    float* __restrict__ out) {
  __shared__ _Float16 ldsA[2][TM * LDA];   // 2 x 10240 B
  __shared__ _Float16 ldsB[2][TN * LDB];   // 2 x  5120 B

  const int tid  = threadIdx.x;
  const int lane = tid & 31;
  const int w    = tid >> 5;                 // wave 0..7 -> M sub-tile
  const size_t m0 = (size_t)blockIdx.y * TM;
  const int n0 = blockIdx.x * TN;

  v8f acc[4];
  #pragma unroll
  for (int i = 0; i < 4; ++i)
    #pragma unroll
    for (int j = 0; j < 8; ++j) acc[i][j] = 0.0f;

  LOAD_TILES(0, 0);
  __syncthreads();

  for (int kk = 0; kk < KTILES; ++kk) {
    const int buf = kk & 1;
    if (kk + 1 < KTILES) LOAD_TILES(kk + 1, buf ^ 1);
    if (kk + 2 < KTILES)                      // gfx1250 global_prefetch_b8
      __builtin_prefetch(Ah + (m0 + (tid >> 1)) * KEXT + (kk + 2) * 32 + (tid & 1) * 16, 0, 1);

    // --- A fragment (16x32 f16): lane half selects K 0-15 vs 16-31 ---
    union { v16h v; uint32_t u[8]; } af;
    {
      const uint32_t* lA = (const uint32_t*)(&ldsA[buf][(w * 16 + (lane & 15)) * LDA]);
      const int hb = (lane >> 4) * 8;        // dword offset = Kbase/2
      #pragma unroll
      for (int j = 0; j < 4; ++j) { af.u[j] = lA[hb + j]; af.u[4 + j] = lA[hb + 4 + j]; }
    }

    // --- 4 B fragments (32x16 f16): lane half selects K 0-15 vs 16-31, col = lane%16 ---
    #pragma unroll
    for (int nt = 0; nt < 4; ++nt) {
      union { v16h v; uint32_t u[8]; } bf;
      const uint32_t* lB = (const uint32_t*)(&ldsB[buf][(nt * 16 + (lane & 15)) * LDB]);
      const int kb = (lane >> 4) * 8;        // dword offset = 16 halves / 2
      #pragma unroll
      for (int j = 0; j < 8; ++j) bf.u[j] = lB[kb + j];
      acc[nt] = __builtin_amdgcn_wmma_f32_16x16x32_f16(
          false, af.v, false, bf.v, (short)0, acc[nt], false, false);
    }
    __syncthreads();   // single barrier: covers buf reads done + buf^1 writes visible
  }

  // --- epilogue: C/D layout -> VGPR r holds row r (lanes 0-15) / r+8 (lanes 16-31) ---
  const size_t mbase = m0 + w * 16 + ((lane >> 4) * 8);
  const int ncol = lane & 15;
  #pragma unroll
  for (int nt = 0; nt < 4; ++nt)
    #pragma unroll
    for (int r = 0; r < 8; ++r)
      out[(mbase + r) * DOUT + n0 + nt * 16 + ncol] = acc[nt][r];
}

extern "C" void kernel_launch(void* const* d_in, const int* in_sizes, int n_in,
                              void* d_out, int out_size, void* d_ws, size_t ws_size,
                              hipStream_t stream) {
  (void)in_sizes; (void)n_in; (void)out_size; (void)ws_size;
  const float* x  = (const float*)d_in[0];   // [256,512,1]
  const float* A  = (const float*)d_in[1];   // [512]
  const float* Bv = (const float*)d_in[2];   // [1,512]
  const float* C  = (const float*)d_in[3];   // [512,512]
  const float* M  = (const float*)d_in[4];   // [512,1,5]
  const float* h0 = (const float*)d_in[5];   // [512]
  float* out = (float*)d_out;                // [256,512,512] fp32

  _Float16* Ah = (_Float16*)d_ws;                                   // 131072 x 544 f16
  const size_t ahBytes = (size_t)BSZ * SEQ * KEXT * sizeof(_Float16);
  _Float16* Bt = (_Float16*)((char*)d_ws + ((ahBytes + 255) & ~(size_t)255)); // 512 x 544 f16 (transposed)

  pack_b<<<(DOUT * KEXT + 255) / 256, 256, 0, stream>>>(C, M, Bt);
  scan_pack_a<<<BSZ, KEXT, 0, stream>>>(x, A, Bv, h0, Ah);
  gemm_wmma<<<dim3(DOUT / TN, (BSZ * SEQ) / TM), 256, 0, stream>>>(Ah, Bt, out);
}
